// AttentionMiRNATAR_38620345926061
// MI455X (gfx1250) — compile-verified, matrix operand order checked by reading, not run on hardware
//
#include <hip/hip_runtime.h>
#include <hip/hip_bf16.h>

typedef __attribute__((ext_vector_type(16))) _Float16 v16h;
typedef __attribute__((ext_vector_type(8)))  float    v8f;

#define LMI 30
#define LLN 2000
#define NB  32

// ---------------------------------------------------------------------------
// Weight pack: f32 [O,I,K] -> f16 [K][O][I]  (B-matrix rows contiguous in k)
// ---------------------------------------------------------------------------
__global__ void pack_w_kernel(const float* __restrict__ w, _Float16* __restrict__ wp,
                              int O, int I, int K) {
  int idx = blockIdx.x * 256 + threadIdx.x;
  int n = O * I * K;
  if (idx >= n) return;
  int t = idx / (O * I);
  int rem = idx - t * O * I;
  int o = rem / I;
  int i = rem - o * I;
  wp[idx] = (_Float16)w[(o * I + i) * K + t];
}

// ---------------------------------------------------------------------------
// Embedding gather -> f16 [R, 128]
// ---------------------------------------------------------------------------
__global__ void embed_kernel(const int* __restrict__ tok, const float* __restrict__ emb,
                             _Float16* __restrict__ out, int R) {
  int r = blockIdx.x;
  int e = threadIdx.x;                       // blockDim.x == 128
  out[(size_t)r * 128 + e] = (_Float16)emb[(size_t)tok[r] * 128 + e];
}

// ---------------------------------------------------------------------------
// WMMA GEMM / tap-summed conv. One wave computes a 16-row x (16*NT)-col tile:
//   Y[r, ocol+n] = act( sum_t sum_k X[r + (t-pad), k] * Wp[t][n][k] + bias[n] )
// Out-of-range / conv-padding rows read from a zeroed scratch line (Zb), so
// the k-loop is branch-free: A fragment loads feed NT back-to-back WMMAs.
// act: 0=none, 1=relu, 2=leaky_relu(0.01)
// ---------------------------------------------------------------------------
template <int NT>
__global__ void wmma_gemm_kernel(const _Float16* __restrict__ X,
                                 const _Float16* __restrict__ Wp,
                                 const float* __restrict__ bias,
                                 _Float16* __restrict__ Y,
                                 const _Float16* __restrict__ Zb,
                                 int R, int L, int Cin, int Cout, int KT,
                                 int ostride, int ocol, int act) {
  int lane  = threadIdx.x;
  int half_ = lane >> 4;
  int lm    = lane & 15;
  int r0    = blockIdx.x * 16;
  int rr    = r0 + lm;
  bool rok  = rr < R;
  int l     = rok ? (rr % L) : 0;
  int nbase = blockIdx.y * 16 * NT + lm;

  v8f acc[NT];
#pragma unroll
  for (int q = 0; q < NT; ++q) acc[q] = (v8f){};

  int pad = (KT - 1) >> 1;
  for (int t = 0; t < KT; ++t) {
    int dt = t - pad;
    bool aok = rok && (l + dt >= 0) && (l + dt < L);
    // branch-free A addressing: padded rows read zeros from Zb
    const _Float16* ab = (aok ? (X + (size_t)(rr + dt) * Cin) : Zb)
                         + (size_t)half_ * 16;
    if (t + 1 < KT) __builtin_prefetch(ab + Cin, 0, 1);   // next tap's A line
    const _Float16* wt = Wp + (size_t)t * Cout * Cin + (size_t)half_ * 16;
    for (int k0 = 0; k0 < Cin; k0 += 32) {
      v16h av = *(const v16h*)(ab + k0);
#pragma unroll
      for (int q = 0; q < NT; ++q) {
        v16h bv = *(const v16h*)(wt + (size_t)(nbase + q * 16) * Cin + k0);
        acc[q] = __builtin_amdgcn_wmma_f32_16x16x32_f16(
            false, av, false, bv, (short)0, acc[q], false, false);
      }
    }
  }

#pragma unroll
  for (int q = 0; q < NT; ++q) {
    int n = nbase + q * 16;
    float bs = bias[n];
#pragma unroll
    for (int v = 0; v < 8; ++v) {
      int m = half_ * 8 + v;         // C/D layout: lanes 0-15 -> M=v, 16-31 -> M=v+8
      int orow = r0 + m;
      if (orow < R) {
        float val = acc[q][v] + bs;
        if (act == 1)      val = fmaxf(val, 0.0f);
        else if (act == 2) val = val > 0.0f ? val : 0.01f * val;
        Y[(size_t)orow * ostride + ocol + n] = (_Float16)val;
      }
    }
  }
}

// ---------------------------------------------------------------------------
// Training-mode BatchNorm (batch statistics), two kernels
// ---------------------------------------------------------------------------
__global__ void bn_stats_kernel(const _Float16* __restrict__ x, int rows, int stride,
                                float* __restrict__ mean, float* __restrict__ rstd) {
  __shared__ float ss[256], sq[256];
  int c = blockIdx.x, t = threadIdx.x;
  float a = 0.f, b = 0.f;
  for (int r = t; r < rows; r += 256) {
    float v = (float)x[(size_t)r * stride + c];
    a += v; b += v * v;
  }
  ss[t] = a; sq[t] = b; __syncthreads();
  for (int s = 128; s > 0; s >>= 1) {
    if (t < s) { ss[t] += ss[t + s]; sq[t] += sq[t + s]; }
    __syncthreads();
  }
  if (t == 0) {
    float m = ss[0] / rows;
    float v = sq[0] / rows - m * m;
    mean[c] = m;
    rstd[c] = rsqrtf(fmaxf(v, 0.f) + 1e-5f);
  }
}

__global__ void bn_apply_kernel(_Float16* __restrict__ x, int rows, int C, int stride,
                                const float* __restrict__ mean, const float* __restrict__ rstd,
                                const float* __restrict__ g, const float* __restrict__ be) {
  int idx = blockIdx.x * 256 + threadIdx.x;
  if (idx >= rows * C) return;
  int r = idx / C, c = idx - r * C;
  size_t o = (size_t)r * stride + c;
  float v = (float)x[o];
  x[o] = (_Float16)((v - mean[c]) * rstd[c] * g[c] + be[c]);
}

// ---------------------------------------------------------------------------
// Pairwise co-attention reduce:
//   S1[b,i,f] = sum_j relu(a[b,i,f] + bb[b,j,f])   (LDS + global atomic acc)
//   S2[b,j,f] = sum_i relu(a[b,i,f] + bb[b,j,f])   (direct store)
// grid (B, LL/jPB), block (64, 4). mi_att tile cached in LDS.
// ---------------------------------------------------------------------------
__global__ void coatt_kernel(const _Float16* __restrict__ amat, const _Float16* __restrict__ bmat,
                             float* __restrict__ S1, float* __restrict__ S2, int LL, int jPB) {
  __shared__ float a_sh[LMI * 64];
  __shared__ float s1_sh[LMI * 64];
  int b = blockIdx.x;
  int tx = threadIdx.x, ty = threadIdx.y;
  int tid = ty * 64 + tx;
  for (int idx = tid; idx < LMI * 64; idx += 256) {
    a_sh[idx]  = (float)amat[(size_t)b * LMI * 64 + idx];
    s1_sh[idx] = 0.f;
  }
  __syncthreads();

  float s1p[LMI];
#pragma unroll
  for (int i = 0; i < LMI; ++i) s1p[i] = 0.f;

  int jbase = blockIdx.y * jPB;
  for (int jj = ty; jj < jPB; jj += 4) {
    int j = jbase + jj;
    float bv = (float)bmat[((size_t)b * LL + j) * 64 + tx];
    float s2a = 0.f;
#pragma unroll
    for (int i = 0; i < LMI; ++i) {
      float v = a_sh[i * 64 + tx] + bv;
      v = fmaxf(v, 0.f);
      s2a += v;
      s1p[i] += v;
    }
    S2[((size_t)b * LL + j) * 64 + tx] = s2a;
  }
#pragma unroll
  for (int i = 0; i < LMI; ++i) atomicAdd(&s1_sh[i * 64 + tx], s1p[i]);
  __syncthreads();
  for (int idx = tid; idx < LMI * 64; idx += 256)
    atomicAdd(&S1[(size_t)b * LMI * 64 + idx], s1_sh[idx]);
}

// f32 -> f16 with scale (used for S1/LL, S2/LM means, feat conversion)
__global__ void cvt_f32_f16_kernel(const float* __restrict__ in, _Float16* __restrict__ out,
                                   int n, float scale) {
  int i = blockIdx.x * 256 + threadIdx.x;
  if (i < n) out[i] = (_Float16)(in[i] * scale);
}

// Attention-weighted global max pool:
//   feat[b, foff+f] = max_l enc[b,l,f] * (sigmoid(att[b,l,f]) + 0.5)
__global__ void pool_kernel(const _Float16* __restrict__ enc, const _Float16* __restrict__ attv,
                            float* __restrict__ feat, int L, int foff) {
  int b = blockIdx.x, f = threadIdx.x;    // blockDim.x == 64
  float m = -3.4e38f;
  for (int l = 0; l < L; ++l) {
    size_t o = ((size_t)b * L + l) * 64 + f;
    float e = (float)enc[o];
    float a = (float)attv[o];
    float s = 1.f / (1.f + __expf(-a));
    m = fmaxf(m, e * (s + 0.5f));
  }
  feat[b * 128 + foff + f] = m;
}

// Final 256 -> 2 projection, f32 output [32,2]
__global__ void head_out_kernel(const _Float16* __restrict__ h3, const float* __restrict__ w,
                                const float* __restrict__ bias, float* __restrict__ out) {
  int t = threadIdx.x;        // 64 threads
  int b = t >> 1, o = t & 1;
  float s = bias[o];
  for (int k = 0; k < 256; ++k) s += (float)h3[b * 256 + k] * w[o * 256 + k];
  out[b * 2 + o] = s;
}

// ---------------------------------------------------------------------------
// Host orchestration
// ---------------------------------------------------------------------------
extern "C" void kernel_launch(void* const* d_in, const int* in_sizes, int n_in,
                              void* d_out, int out_size, void* d_ws, size_t ws_size,
                              hipStream_t stream) {
  (void)in_sizes; (void)n_in; (void)out_size; (void)ws_size;
  // d_in flattening assumption (insertion order):
  // 0 mi, 1 lnc, 2 strmi(unused), 3 strlnc(unused), 4 emb1, 5 emb2,
  // 6/7 mic1.w/b, 8/9 mic2.w/b, 10..37 mi_ms (7 conv layers x (w,b,g,be)),
  // 38/39 lc1, 40/41 lc2, 42..69 lnc_ms, 70/71 matt, 72/73 latt, 74/75 att,
  // 76/77 bn7, 78/79 bn8, 80/81 bn9, 82..89 fc1,fc2,fc3,out (w,b each)
  const int* mi  = (const int*)d_in[0];
  const int* lnc = (const int*)d_in[1];
  auto P = [&](int i) { return (const float*)d_in[i]; };
  char* ws = (char*)d_ws;

  // ---- workspace allocator (deterministic, 256B aligned) ----
  size_t cur = 0;
  auto alloc = [&](size_t bytes) {
    size_t o = cur;
    cur += (bytes + 255) & ~(size_t)255;
    return o;
  };

  // ---- packed f16 weights ----
  static const int NP = 24;
  static const int pIdx[NP] = { 6,  8, 10, 14, 18, 22, 26, 30, 34, 38, 40, 42,
                               46, 50, 54, 58, 62, 66, 70, 72, 74, 82, 84, 86};
  static const int pO[NP]   = {64,128, 16, 64, 16, 64, 16, 64, 16, 64,128, 16,
                               64, 16, 64, 16, 64, 16, 64, 64, 64,512,512,256};
  static const int pI[NP]   = {128,64,128,128, 64,128, 64,128, 64,128, 64,128,
                              128, 64,128, 64,128, 64, 64, 64, 64,128,512,512};
  static const int pK[NP]   = { 1,  3,  1,  1,  3,  1,  5,  1,  7,  1,  3,  1,
                                1,  3,  1,  5,  1,  7,  1,  1,  1,  1,  1,  1};
  size_t wpOff[96];
  for (int q = 0; q < NP; ++q)
    wpOff[pIdx[q]] = alloc((size_t)pO[q] * pI[q] * pK[q] * 2);
  auto WP = [&](int i) { return (_Float16*)(ws + wpOff[i]); };

  // ---- activation buffers (aggressive aliasing; all dead-before-reuse) ----
  const int Rmi = NB * LMI, Rln = NB * LLN;
  size_t ZERO   = alloc(1024);                      // zero line for conv padding
  size_t A_mi   = alloc((size_t)Rmi * 128 * 2);     // x0/x2 (mi)
  size_t B_mi   = alloc((size_t)Rmi * 64 * 2);      // x1 / branch temp (mi)
  size_t ENC_mi = alloc((size_t)Rmi * 64 * 2);
  size_t ATT_mi = alloc((size_t)Rmi * 64 * 2);
  size_t A_ln   = alloc((size_t)Rln * 128 * 2);     // x0/x2 (lnc); later S2 f32 view
  size_t B_ln   = alloc((size_t)Rln * 64 * 2);      // x1 / branch temp; later S2 f16
  size_t ENC_ln = alloc((size_t)Rln * 64 * 2);
  size_t ATT_ln = alloc((size_t)Rln * 64 * 2);      // later att2_lnc
  size_t S1     = alloc((size_t)Rmi * 64 * 4);
  size_t S1H    = alloc((size_t)Rmi * 64 * 2);
  size_t AT2_mi = alloc((size_t)Rmi * 64 * 2);
  size_t FEAT   = alloc(32 * 128 * 4);
  size_t FEATH  = alloc(32 * 128 * 2);
  size_t H1     = alloc(32 * 512 * 2);
  size_t H2     = alloc(32 * 512 * 2);
  size_t H3     = alloc(32 * 256 * 2);
  size_t BNM    = alloc(512 * 4);
  size_t BNR    = alloc(512 * 4);
  size_t S2  = A_ln;     // f32, exactly Rln*64*4 == Rln*128*2 bytes
  size_t S2H = B_ln;     // f16
  size_t AT2_ln = ATT_ln;

  auto H = [&](size_t o) { return (_Float16*)(ws + o); };
  auto Fp = [&](size_t o) { return (float*)(ws + o); };
  const _Float16* Zb = H(ZERO);

  // zero scratch line (read as conv padding) — stream-ordered before all gemms
  hipMemsetAsync(ws + ZERO, 0, 1024, stream);

  // ---- pack all GEMM weights to f16 ----
  for (int q = 0; q < NP; ++q) {
    int n = pO[q] * pI[q] * pK[q];
    pack_w_kernel<<<(n + 255) / 256, 256, 0, stream>>>(P(pIdx[q]), WP(pIdx[q]),
                                                       pO[q], pI[q], pK[q]);
  }

  // ---- helpers ----
  auto gemm = [&](const _Float16* X, int Cin, const _Float16* Wq, const float* bias,
                  _Float16* Y, int Cout, int KT, int ostride, int ocol, int act,
                  int R, int L) {
    if (Cout % 64 == 0) {
      dim3 g((R + 15) / 16, Cout / 64);
      wmma_gemm_kernel<4><<<g, 32, 0, stream>>>(X, Wq, bias, Y, Zb, R, L, Cin,
                                                Cout, KT, ostride, ocol, act);
    } else {
      dim3 g((R + 15) / 16, Cout / 16);
      wmma_gemm_kernel<1><<<g, 32, 0, stream>>>(X, Wq, bias, Y, Zb, R, L, Cin,
                                                Cout, KT, ostride, ocol, act);
    }
  };
  auto bn = [&](_Float16* x, int rows, int C, int stride, const float* g, const float* be) {
    bn_stats_kernel<<<C, 256, 0, stream>>>(x, rows, stride, Fp(BNM), Fp(BNR));
    bn_apply_kernel<<<(rows * C + 255) / 256, 256, 0, stream>>>(x, rows, C, stride,
                                                                Fp(BNM), Fp(BNR), g, be);
  };

  // ---- encoder: embed -> c1(1x1) -> c2(k3) -> 4-branch MSCNN -> att linear ----
  auto run_encoder = [&](const int* tok, int R, int L, const float* embw,
                         int ic1, int ic2, int ms0, int iatt,
                         _Float16* A, _Float16* Bt, _Float16* ENC, _Float16* ATT) {
    embed_kernel<<<R, 128, 0, stream>>>(tok, embw, A, R);
    gemm(A, 128, WP(ic1), P(ic1 + 1), Bt, 64, 1, 64, 0, 0, R, L);      // c1: E->F
    gemm(Bt, 64, WP(ic2), P(ic2 + 1), A, 128, 3, 128, 0, 0, R, L);     // c2: F->E, k3
    // branch 0: 1x1 E->16, relu, BN  -> enc[:, 0:16)
    gemm(A, 128, WP(ms0), P(ms0 + 1), ENC, 16, 1, 64, 0, 1, R, L);
    bn(ENC, R, 16, 64, P(ms0 + 2), P(ms0 + 3));
    // branches 1..3: (1x1 E->F, relu, BN) -> (k{3,5,7} F->16, relu, BN)
    const int ktap[3] = {3, 5, 7};
    for (int br = 0; br < 3; ++br) {
      int c1i = ms0 + 4 + br * 8, c2i = c1i + 4;
      gemm(A, 128, WP(c1i), P(c1i + 1), Bt, 64, 1, 64, 0, 1, R, L);
      bn(Bt, R, 64, 64, P(c1i + 2), P(c1i + 3));
      gemm(Bt, 64, WP(c2i), P(c2i + 1), ENC, 16, ktap[br], 64, 16 * (br + 1), 1, R, L);
      bn(ENC + 16 * (br + 1), R, 16, 64, P(c2i + 2), P(c2i + 3));
    }
    gemm(ENC, 64, WP(iatt), P(iatt + 1), ATT, 64, 1, 64, 0, 0, R, L);  // att proj
  };

  run_encoder(mi,  Rmi, LMI, P(4), 6, 8, 10, 70, H(A_mi), H(B_mi), H(ENC_mi), H(ATT_mi));
  run_encoder(lnc, Rln, LLN, P(5), 38, 40, 42, 72, H(A_ln), H(B_ln), H(ENC_ln), H(ATT_ln));

  // ---- pairwise co-attention (relu-sum pulled out of the att Linear) ----
  hipMemsetAsync(ws + S1, 0, (size_t)Rmi * 64 * 4, stream);
  coatt_kernel<<<dim3(NB, 10), dim3(64, 4), 0, stream>>>(H(ATT_mi), H(ATT_ln),
                                                         Fp(S1), Fp(S2), LLN, 200);
  cvt_f32_f16_kernel<<<(Rmi * 64 + 255) / 256, 256, 0, stream>>>(Fp(S1), H(S1H),
                                                                 Rmi * 64, 1.0f / LLN);
  cvt_f32_f16_kernel<<<(Rln * 64 + 255) / 256, 256, 0, stream>>>(Fp(S2), H(S2H),
                                                                 Rln * 64, 1.0f / LMI);
  gemm(H(S1H), 64, WP(74), P(75), H(AT2_mi), 64, 1, 64, 0, 0, Rmi, LMI);
  gemm(H(S2H), 64, WP(74), P(75), H(AT2_ln), 64, 1, 64, 0, 0, Rln, LLN);

  // ---- sigmoid-gated max pool -> feat [32,128] ----
  pool_kernel<<<NB, 64, 0, stream>>>(H(ENC_mi), H(AT2_mi), Fp(FEAT), LMI, 0);
  pool_kernel<<<NB, 64, 0, stream>>>(H(ENC_ln), H(AT2_ln), Fp(FEAT), LLN, 64);

  // ---- head: BN7 -> fc1+lrelu+BN8 -> fc2+lrelu+BN9 -> fc3+lrelu -> out ----
  cvt_f32_f16_kernel<<<(32 * 128 + 255) / 256, 256, 0, stream>>>(Fp(FEAT), H(FEATH),
                                                                 32 * 128, 1.0f);
  bn(H(FEATH), 32, 128, 128, P(76), P(77));
  gemm(H(FEATH), 128, WP(82), P(83), H(H1), 512, 1, 512, 0, 2, 32, 32);
  bn(H(H1), 32, 512, 512, P(78), P(79));
  gemm(H(H1), 512, WP(84), P(85), H(H2), 512, 1, 512, 0, 2, 32, 32);
  bn(H(H2), 32, 512, 512, P(80), P(81));
  gemm(H(H2), 512, WP(86), P(87), H(H3), 256, 1, 256, 0, 2, 32, 32);
  head_out_kernel<<<1, 64, 0, stream>>>(H(H3), P(88), P(89), (float*)d_out);
}
